// ImprovedKalmanNet_70901320123083
// MI455X (gfx1250) — compile-verified, matrix operand order checked by reading, not run on hardware
//
#include <hip/hip_runtime.h>

typedef unsigned int u32;
typedef unsigned short u16;
typedef __bf16 bf16_t;
typedef bf16_t v16bf __attribute__((ext_vector_type(16)));
typedef float v8f __attribute__((ext_vector_type(8)));

// ---------------- workspace layout (u32/float units) ----------------
static constexpr int OFF_GSRZ = 0;                       // 512  (gi+gh const, r|z gates)
static constexpr int OFF_GSIN = 512;                     // 256  (gi const, n gate)
static constexpr int OFF_GSHN = 768;                     // 256  (gh const, n gate)
static constexpr int OFF_HS0  = 1024;                    // 256  (prior_S flat)
static constexpr int OFF_H2C  = 1280;                    // 2560 (h2 const part)
static constexpr int OFF_KC   = 3840;                    // 512  (0.5*K_kf + 0.5*b2b)
static constexpr int OFF_FW7  = 4352;                    // 10*1*256
static constexpr int OFF_FWS  = OFF_FW7  + 10*256;       // 48*5*256
static constexpr int OFF_FW2A = OFF_FWS  + 48*5*256;     // 160*8*256
static constexpr int OFF_FW2B = OFF_FW2A + 160*8*256;    // 32*80*256
static constexpr int OFF_TMP_GI  = OFF_FW2B + 32*80*256; // 3072
static constexpr int OFF_TMP_GH  = OFF_TMP_GI + 3072;    // 3072
static constexpr int OFF_TMP_X   = OFF_TMP_GH + 3072;    // 1184 (GRU-Si input)
static constexpr int OFF_TMP_SIG = OFF_TMP_X + 1184;     // 1024 (out_Sigma)
static constexpr int OFF_TMP_O1  = OFF_TMP_SIG + 1024;   // 256  (out1)

// ---------------- helpers ----------------
__device__ __forceinline__ u16 f2bf(float x) {
  u32 u = __float_as_uint(x);
  u32 r = u + 0x7FFFu + ((u >> 16) & 1u);   // round-to-nearest-even
  return (u16)(r >> 16);
}
__device__ __forceinline__ float sigf(float x) { return 1.0f / (1.0f + __expf(-x)); }
__device__ __forceinline__ v8f zero8() { v8f z = {0.f,0.f,0.f,0.f,0.f,0.f,0.f,0.f}; return z; }

union AF { v16bf v; uint4 q[2]; };

// A fragment from LDS row-major bf16 (16B-aligned rows), ktile t.
__device__ __forceinline__ AF load_a(const u16* A, int rowStrideBytes, int t,
                                     int lm, int lh) {
  AF a;
  const char* rp = (const char*)A + lm * rowStrideBytes;
  a.q[0] = *(const uint4*)(rp + t*64 + lh*16);        // K = t*32 + lh*8 .. +7
  a.q[1] = *(const uint4*)(rp + t*64 + 32 + lh*16);   // K = t*32 + 16 + lh*8 .. +7
  return a;
}
// B fragment from pre-packed global (256 dwords per 16x32 tile, lane*8 each).
__device__ __forceinline__ AF load_b(const u32* __restrict__ frag, int idx, int lane) {
  AF b;
  const u32* bp = frag + (size_t)idx * 256 + lane * 8;
  b.q[0] = *(const uint4*)bp;
  b.q[1] = *(const uint4*)(bp + 4);
  return b;
}
__device__ __forceinline__ v8f wmma_bf16(AF a, AF b, v8f c) {
  return __builtin_amdgcn_wmma_f32_16x16x32_bf16(false, a.v, false, b.v,
                                                 (short)0, c, false, false);
}

// ---------------- setup: batch-invariant constants ----------------
extern "C" __global__ __launch_bounds__(256) void kA_gruQ(
    const float* __restrict__ WihQ, const float* __restrict__ WhhQ,
    const float* __restrict__ bihQ, const float* __restrict__ bhhQ,
    const float* __restrict__ b5, const float* __restrict__ pQ, float* wsf) {
  int o = blockIdx.x * blockDim.x + threadIdx.x;
  if (o >= 3072) return;
  float s = bihQ[o];
  const float* w = WihQ + (size_t)o * 160;
  for (int k = 0; k < 160; ++k) s += fmaxf(b5[k], 0.f) * w[k];
  wsf[OFF_TMP_GI + o] = s;
  float t = bhhQ[o];
  const float* wh = WhhQ + (size_t)o * 1024;
  for (int k = 0; k < 1024; ++k) t += pQ[k] * wh[k];
  wsf[OFF_TMP_GH + o] = t;
}

extern "C" __global__ __launch_bounds__(256) void kA_gateQ(
    const float* __restrict__ pQ, const float* __restrict__ b6, float* wsf) {
  int c = blockIdx.x * blockDim.x + threadIdx.x;
  if (c < 1024) {
    float r = sigf(wsf[OFF_TMP_GI + c] + wsf[OFF_TMP_GH + c]);
    float z = sigf(wsf[OFF_TMP_GI + 1024 + c] + wsf[OFF_TMP_GH + 1024 + c]);
    float n = tanhf(wsf[OFF_TMP_GI + 2048 + c] + r * wsf[OFF_TMP_GH + 2048 + c]);
    wsf[OFF_TMP_X + c] = (1.f - z) * n + z * pQ[c];
  }
  if (c < 160) wsf[OFF_TMP_X + 1024 + c] = fmaxf(b6[c], 0.f);
}

extern "C" __global__ __launch_bounds__(256) void kA_gruSi(
    const float* __restrict__ WihSi, const float* __restrict__ WhhSi,
    const float* __restrict__ bihSi, const float* __restrict__ bhhSi,
    const float* __restrict__ pSig, float* wsf) {
  int o = blockIdx.x * blockDim.x + threadIdx.x;
  if (o >= 3072) return;
  float s = bihSi[o];
  const float* w = WihSi + (size_t)o * 1184;
  const float* x = wsf + OFF_TMP_X;
  for (int k = 0; k < 1184; ++k) s += x[k] * w[k];
  wsf[OFF_TMP_GI + o] = s;
  float t = bhhSi[o];
  const float* wh = WhhSi + (size_t)o * 1024;
  for (int k = 0; k < 1024; ++k) t += pSig[k] * wh[k];
  wsf[OFF_TMP_GH + o] = t;
}

extern "C" __global__ __launch_bounds__(256) void kA_gateSi(
    const float* __restrict__ pSig, float* wsf) {
  int c = blockIdx.x * blockDim.x + threadIdx.x;
  if (c >= 1024) return;
  float r = sigf(wsf[OFF_TMP_GI + c] + wsf[OFF_TMP_GH + c]);
  float z = sigf(wsf[OFF_TMP_GI + 1024 + c] + wsf[OFF_TMP_GH + 1024 + c]);
  float n = tanhf(wsf[OFF_TMP_GI + 2048 + c] + r * wsf[OFF_TMP_GH + 2048 + c]);
  wsf[OFF_TMP_SIG + c] = (1.f - z) * n + z * pSig[c];
}

extern "C" __global__ __launch_bounds__(256) void kA_out1(
    const float* __restrict__ W1, const float* __restrict__ b1, float* wsf) {
  int o = blockIdx.x * blockDim.x + threadIdx.x;
  if (o >= 256) return;
  float s = b1[o];
  const float* w = W1 + (size_t)o * 1024;
  const float* sg = wsf + OFF_TMP_SIG;
  for (int k = 0; k < 1024; ++k) s += sg[k] * w[k];
  wsf[OFF_TMP_O1 + o] = fmaxf(s, 0.f);
}

extern "C" __global__ __launch_bounds__(256) void kA_mix(
    const float* __restrict__ WihS, const float* __restrict__ WhhS,
    const float* __restrict__ bihS, const float* __restrict__ bhhS,
    const float* __restrict__ W2a, const float* __restrict__ b2a,
    const float* __restrict__ pS, float* wsf) {
  int o = blockIdx.x * blockDim.x + threadIdx.x;
  if (o < 768) {
    float s = bihS[o];
    const float* w = WihS + (size_t)o * 416;
    const float* o1 = wsf + OFF_TMP_O1;
    for (int k = 0; k < 256; ++k) s += o1[k] * w[k];
    float t = bhhS[o];
    const float* wh = WhhS + (size_t)o * 256;
    for (int k = 0; k < 256; ++k) t += pS[k] * wh[k];
    if (o < 512) wsf[OFF_GSRZ + o] = s + t;
    else { wsf[OFF_GSIN + o - 512] = s; wsf[OFF_GSHN + o - 512] = t; }
  } else if (o < 768 + 2560) {
    int o2 = o - 768;
    float s = b2a[o2];
    const float* w = W2a + (size_t)o2 * 1280;
    const float* sg = wsf + OFF_TMP_SIG;
    for (int k = 0; k < 1024; ++k) s += sg[k] * w[k];
    wsf[OFF_H2C + o2] = s;
  } else if (o < 768 + 2560 + 256) {
    int c = o - 768 - 2560;
    wsf[OFF_HS0 + c] = pS[c];
  }
}

// Analytic KF gain (constant across batch): Gauss-Jordan on SPD S.
extern "C" __global__ __launch_bounds__(256) void kA_kf(
    const float* __restrict__ Fm, const float* __restrict__ Hm,
    const float* __restrict__ pQ, const float* __restrict__ pSig,
    const float* __restrict__ pS, const float* __restrict__ b2b, float* wsf) {
  __shared__ float T1[1024], PP[1024], T2[512], PHt[512], Smat[256], aug[16 * 48];
  int tid = threadIdx.x;
  for (int idx = tid; idx < 1024; idx += 256) {
    int i = idx >> 5, k = idx & 31;
    float s = 0.f;
    for (int j = 0; j < 32; ++j) s += Fm[i*32 + j] * pSig[j*32 + k];
    T1[idx] = s;
  }
  __syncthreads();
  for (int idx = tid; idx < 1024; idx += 256) {
    int i = idx >> 5, l = idx & 31;
    float s = pQ[idx];
    for (int k = 0; k < 32; ++k) s += T1[i*32 + k] * Fm[l*32 + k];
    PP[idx] = s;
  }
  __syncthreads();
  for (int idx = tid; idx < 512; idx += 256) {
    int i = idx >> 5, k = idx & 31;
    float s = 0.f;
    for (int j = 0; j < 32; ++j) s += Hm[i*32 + j] * PP[j*32 + k];
    T2[idx] = s;
  }
  for (int idx = tid; idx < 512; idx += 256) {
    int i = idx >> 4, jn = idx & 15;
    float s = 0.f;
    for (int k = 0; k < 32; ++k) s += PP[i*32 + k] * Hm[jn*32 + k];
    PHt[idx] = s;
  }
  __syncthreads();
  for (int idx = tid; idx < 256; idx += 256) {
    int i = idx >> 4, l = idx & 15;
    float s = pS[idx];
    for (int k = 0; k < 32; ++k) s += T2[i*32 + k] * Hm[l*32 + k];
    Smat[idx] = s;
  }
  __syncthreads();
  if (tid == 0) {
    for (int r = 0; r < 16; ++r) {
      for (int c = 0; c < 16; ++c) aug[r*48 + c] = Smat[r*16 + c];
      for (int i = 0; i < 32; ++i) aug[r*48 + 16 + i] = PHt[i*16 + r];
    }
    for (int p = 0; p < 16; ++p) {
      float inv = 1.f / aug[p*48 + p];
      for (int c = 0; c < 48; ++c) aug[p*48 + c] *= inv;
      for (int r = 0; r < 16; ++r) if (r != p) {
        float f = aug[r*48 + p];
        for (int c = 0; c < 48; ++c) aug[r*48 + c] -= f * aug[p*48 + c];
      }
    }
  }
  __syncthreads();
  for (int idx = tid; idx < 512; idx += 256) {
    int i = idx >> 4, jn = idx & 15;
    wsf[OFF_KC + idx] = 0.5f * aug[jn*48 + 16 + i] + 0.5f * b2b[idx];
  }
}

// Pack W (row-major [O, Kdim], K-slice at koff, optional scale) into WMMA
// B-fragments: B[k,n] = scale * W[nt*16+n, koff+k]; frag f = nt*nKT+kt.
extern "C" __global__ __launch_bounds__(256) void knet_pack(
    const float* __restrict__ W, int Kdim, int koff, int nNT, int nKT,
    float scale, u32* __restrict__ dst) {
  int total = nNT * nKT * 256;
  for (int idx = blockIdx.x * blockDim.x + threadIdx.x; idx < total;
       idx += gridDim.x * blockDim.x) {
    int f = idx >> 8, r = idx & 255;
    int nt = f / nKT, kt = f % nKT;
    int lane = r >> 3, v = r & 7;
    int n = lane & 15, h = lane >> 4;
    int kin = (v < 4) ? (h*8 + 2*v) : (16 + h*8 + 2*(v - 4));
    int k = kt * 32 + kin;
    const float* wp = W + (size_t)(nt*16 + n) * Kdim + koff + k;
    u32 lo = f2bf(wp[0] * scale);
    u32 hi = f2bf(wp[1] * scale);
    dst[idx] = lo | (hi << 16);
  }
}

// ---------------- main fused per-batch kernel: 32 rows / block ----------------
extern "C" __global__ __launch_bounds__(256) void knet_main(
    const float* __restrict__ y, const float* __restrict__ x0,
    const float* __restrict__ Fm, const float* __restrict__ Hm,
    const float* __restrict__ b7,
    const float* __restrict__ cst, const u32* __restrict__ frg,
    float* __restrict__ out) {
  __shared__ float s_x0[2][512];                                 // 16x32 f32
  __shared__ float s_mxp[2][512];                                // 16x32 m1x_prior
  __shared__ float s_res[2][256];                                // 16x16 y - m1y
  __shared__ u16 s_in7[2][512] __attribute__((aligned(16)));     // 16x32 bf16
  __shared__ u16 s_o7[2][2560] __attribute__((aligned(16)));     // 16x160 bf16
  __shared__ u16 s_oS[2][4096] __attribute__((aligned(16)));     // 16x256 bf16
  __shared__ u16 s_h2[2][4096] __attribute__((aligned(16)));     // 16x256 bf16 chunk

  const int tid = threadIdx.x;
  const int b0 = blockIdx.x * 32;
  const int wave = tid >> 5, lane = tid & 31;
  const int lm = lane & 15, lh = lane >> 4;

  const u32* fW7  = frg + OFF_FW7;
  const u32* fWS  = frg + OFF_FWS;
  const u32* fW2a = frg + OFF_FW2A;
  const u32* fW2b = frg + OFF_FW2B;
  const float* gS_rz = cst + OFF_GSRZ;
  const float* gS_in = cst + OFF_GSIN;
  const float* gS_hn = cst + OFF_GSHN;
  const float* hS0   = cst + OFF_HS0;
  const float* h2c   = cst + OFF_H2C;
  const float* Kc    = cst + OFF_KC;

  // ---- stage 0: prior / innovation / normalize, both row tiles ----
  {
    int m = tid >> 4, j = tid & 15;
    #pragma unroll
    for (int rt = 0; rt < 2; ++rt) {
      size_t gm = (size_t)(b0 + rt*16 + m);
      s_x0[rt][m*32 + j]      = x0[gm*32 + j];
      s_x0[rt][m*32 + 16 + j] = x0[gm*32 + 16 + j];
    }
    __syncthreads();
    #pragma unroll
    for (int rt = 0; rt < 2; ++rt) {
      float a0 = 0.f, a1 = 0.f;
      for (int k = 0; k < 32; ++k) {
        float xv = s_x0[rt][m*32 + k];
        a0 += Fm[j*32 + k] * xv;
        a1 += Fm[(j + 16)*32 + k] * xv;
      }
      s_mxp[rt][m*32 + j] = a0;
      s_mxp[rt][m*32 + 16 + j] = a1;
    }
    __syncthreads();
    #pragma unroll
    for (int rt = 0; rt < 2; ++rt) {
      float yv = y[(size_t)(b0 + rt*16 + m)*16 + j];
      float yp = 0.f, my = 0.f;
      for (int k = 0; k < 32; ++k) {
        float hv = Hm[j*32 + k];
        yp += hv * s_x0[rt][m*32 + k];
        my += hv * s_mxp[rt][m*32 + k];
      }
      float d = yv - yp, e = yv - my;
      s_res[rt][m*16 + j] = e;
      float nd = d*d, ne = e*e;
      for (int s = 1; s < 16; s <<= 1) { nd += __shfl_xor(nd, s, 32); ne += __shfl_xor(ne, s, 32); }
      nd = fmaxf(sqrtf(nd), 1e-12f);
      ne = fmaxf(sqrtf(ne), 1e-12f);
      s_in7[rt][m*32 + j]      = f2bf(d / nd);
      s_in7[rt][m*32 + 16 + j] = f2bf(e / ne);
    }
  }
  __syncthreads();

  // ---- stage 1: out7 = relu(in7 @ W7^T + b7), 2 x 16x160; B reused across tiles ----
  for (int nt = wave; nt < 10; nt += 8) {
    AF b = load_b(fW7, nt, lane);
    int cb = nt*16 + lm;
    float bias = b7[cb];
    #pragma unroll
    for (int rt = 0; rt < 2; ++rt) {
      AF a = load_a(s_in7[rt], 64, 0, lm, lh);
      v8f c = wmma_bf16(a, b, zero8());
      #pragma unroll
      for (int v = 0; v < 8; ++v)
        s_o7[rt][(v + 8*lh)*160 + cb] = f2bf(fmaxf(c[v] + bias, 0.f));
    }
  }
  __syncthreads();

  // ---- stage 2: GRU-S variable part + gates -> out_S (2 x 16x256) ----
  {
    v8f a6[6][2];
    #pragma unroll
    for (int g = 0; g < 6; ++g) { a6[g][0] = zero8(); a6[g][1] = zero8(); }
    for (int t = 0; t < 5; ++t) {
      AF a0 = load_a(s_o7[0], 320, t, lm, lh);
      AF a1 = load_a(s_o7[1], 320, t, lm, lh);
      #pragma unroll
      for (int g = 0; g < 6; ++g) {
        AF b = load_b(fWS, (wave + 8*g)*5 + t, lane);
        a6[g][0] = wmma_bf16(a0, b, a6[g][0]);
        a6[g][1] = wmma_bf16(a1, b, a6[g][1]);
      }
    }
    #pragma unroll
    for (int q = 0; q < 2; ++q) {
      int ct = wave + 8*q, cb = ct*16 + lm;
      float crr = gS_rz[cb], crz = gS_rz[256 + cb];
      float cin = gS_in[cb], chn = gS_hn[cb], h0 = hS0[cb];
      #pragma unroll
      for (int rt = 0; rt < 2; ++rt) {
        #pragma unroll
        for (int v = 0; v < 8; ++v) {
          float r = sigf(a6[q][rt][v] + crr);
          float z = sigf(a6[2 + q][rt][v] + crz);
          float n = tanhf(a6[4 + q][rt][v] + cin + r * chn);
          s_oS[rt][(v + 8*lh)*256 + cb] = f2bf((1.f - z) * n + z * h0);
        }
      }
    }
  }
  __syncthreads();

  // ---- stages 3+4 chunked: h2 chunk (2 x 16x256) -> 0.5*K_nn accumulators ----
  v8f k4[4][2];
  #pragma unroll
  for (int q = 0; q < 4; ++q) { k4[q][0] = zero8(); k4[q][1] = zero8(); }

  for (int ch = 0; ch < 10; ++ch) {
    v8f cc[2][2];
    #pragma unroll
    for (int n = 0; n < 2; ++n) { cc[n][0] = zero8(); cc[n][1] = zero8(); }
    // stage 3: h2 columns [ch*256, ch*256+256)
    for (int t = 0; t < 8; ++t) {
      AF a0 = load_a(s_oS[0], 512, t, lm, lh);
      AF a1 = load_a(s_oS[1], 512, t, lm, lh);
      #pragma unroll
      for (int n = 0; n < 2; ++n) {
        AF b = load_b(fW2a, (ch*16 + wave*2 + n)*8 + t, lane);
        cc[n][0] = wmma_bf16(a0, b, cc[n][0]);
        cc[n][1] = wmma_bf16(a1, b, cc[n][1]);
      }
    }
    #pragma unroll
    for (int n = 0; n < 2; ++n) {
      int ntl = wave*2 + n, cl = ntl*16 + lm, cg = (ch*16 + ntl)*16 + lm;
      float hc = h2c[cg];
      #pragma unroll
      for (int rt = 0; rt < 2; ++rt) {
        #pragma unroll
        for (int v = 0; v < 8; ++v)
          s_h2[rt][(v + 8*lh)*256 + cl] = f2bf(fmaxf(cc[n][rt][v] + hc, 0.f));
      }
    }
    __syncthreads();
    // stage 4: accumulate 0.5*K_nn over this K chunk (W2b packed with scale 0.5)
    for (int t = 0; t < 8; ++t) {
      AF a0 = load_a(s_h2[0], 512, t, lm, lh);
      AF a1 = load_a(s_h2[1], 512, t, lm, lh);
      #pragma unroll
      for (int q = 0; q < 4; ++q) {
        AF b = load_b(fW2b, (wave + 8*q)*80 + ch*8 + t, lane);
        k4[q][0] = wmma_bf16(a0, b, k4[q][0]);
        k4[q][1] = wmma_bf16(a1, b, k4[q][1]);
      }
    }
    __syncthreads();
  }

  // ---- stage 5: m1x_post = m1x_prior + (Kc + 0.5*K_nn) @ resid ----
  // ntile q maps to state component i = wave + 8q; lane lm is j. Cross-lane dot.
  #pragma unroll
  for (int q = 0; q < 4; ++q) {
    int i = wave + 8*q;
    float kc = Kc[i*16 + lm];
    #pragma unroll
    for (int rt = 0; rt < 2; ++rt) {
      #pragma unroll
      for (int v = 0; v < 8; ++v) {
        int m = v + 8*lh;
        float p = (kc + k4[q][rt][v]) * s_res[rt][m*16 + lm];
        for (int s = 1; s < 16; s <<= 1) p += __shfl_xor(p, s, 32);
        if (lm == v)
          out[(size_t)(b0 + rt*16 + m)*32 + i] = s_mxp[rt][m*32 + i] + p;
      }
    }
  }
}

// ---------------- host launcher ----------------
extern "C" void kernel_launch(void* const* d_in, const int* in_sizes, int n_in,
                              void* d_out, int out_size, void* d_ws, size_t ws_size,
                              hipStream_t stream) {
  const float* y     = (const float*)d_in[0];
  const float* x0    = (const float*)d_in[1];
  const float* Fm    = (const float*)d_in[2];
  const float* Hm    = (const float*)d_in[3];
  const float* pQ    = (const float*)d_in[4];
  const float* pSig  = (const float*)d_in[5];
  const float* pS    = (const float*)d_in[6];
  const float* WihQ  = (const float*)d_in[7];
  const float* WhhQ  = (const float*)d_in[8];
  const float* bihQ  = (const float*)d_in[9];
  const float* bhhQ  = (const float*)d_in[10];
  const float* WihSi = (const float*)d_in[11];
  const float* WhhSi = (const float*)d_in[12];
  const float* bihSi = (const float*)d_in[13];
  const float* bhhSi = (const float*)d_in[14];
  const float* WihS  = (const float*)d_in[15];
  const float* WhhS  = (const float*)d_in[16];
  const float* bihS  = (const float*)d_in[17];
  const float* bhhS  = (const float*)d_in[18];
  const float* W1    = (const float*)d_in[19];
  const float* b1    = (const float*)d_in[20];
  const float* W2a   = (const float*)d_in[21];
  const float* b2a   = (const float*)d_in[22];
  const float* W2b   = (const float*)d_in[23];
  const float* b2b   = (const float*)d_in[24];
  const float* b5    = (const float*)d_in[26];
  const float* b6    = (const float*)d_in[28];
  const float* W7    = (const float*)d_in[29];
  const float* b7    = (const float*)d_in[30];

  float* wsf = (float*)d_ws;
  u32*   wsu = (u32*)d_ws;

  // batch-invariant constants (dependent chain; stream-ordered)
  kA_gruQ <<<12, 256, 0, stream>>>(WihQ, WhhQ, bihQ, bhhQ, b5, pQ, wsf);
  kA_gateQ<<<4,  256, 0, stream>>>(pQ, b6, wsf);
  kA_gruSi<<<12, 256, 0, stream>>>(WihSi, WhhSi, bihSi, bhhSi, pSig, wsf);
  kA_gateSi<<<4, 256, 0, stream>>>(pSig, wsf);
  kA_out1 <<<1,  256, 0, stream>>>(W1, b1, wsf);
  kA_mix  <<<14, 256, 0, stream>>>(WihS, WhhS, bihS, bhhS, W2a, b2a, pS, wsf);
  kA_kf   <<<1,  256, 0, stream>>>(Fm, Hm, pQ, pSig, pS, b2b, wsf);

  // weight packing into WMMA B-fragments (independent of constants).
  // W2b is pre-scaled by 0.5 (hybrid alpha folded into the GEMM).
  knet_pack<<<64,   256, 0, stream>>>(W7,   32,   0,    10,  1,  1.0f, wsu + OFF_FW7);
  knet_pack<<<256,  256, 0, stream>>>(WihS, 416,  256,  48,  5,  1.0f, wsu + OFF_FWS);
  knet_pack<<<1024, 256, 0, stream>>>(W2a,  1280, 1024, 160, 8,  1.0f, wsu + OFF_FW2A);
  knet_pack<<<2048, 256, 0, stream>>>(W2b,  2560, 0,    32,  80, 0.5f, wsu + OFF_FW2B);

  // fused per-batch pipeline: 16384/32 = 512 blocks, 2 row tiles each
  knet_main<<<512, 256, 0, stream>>>(y, x0, Fm, Hm, b7, wsf, wsu, (float*)d_out);
}
